// SparseAttention_74131135529912
// MI455X (gfx1250) — matrix operand order repaired and checked by reading, not compiled, not run on hardware
//
#include <hip/hip_runtime.h>

// ---------------------------------------------------------------------------
// Sparse (sliding-window) attention for MI455X / gfx1250, bf16 WMMA pipeline.
//   stage 1: Q/K/V projections  (f32 in -> bf16 out, head layout, Q pre-scaled)
//   stage 2: banded flash attention (bf16 WMMA scores + PV, f32 online softmax)
//   stage 3: output projection  (bf16 in -> f32 out + bias)
// gfx1250 paths: v_wmma_f32_16x16x32_bf16, global_load_async_to_lds_b128 +
// s_wait_asynccnt, global_prefetch_b8, wave32 cross-lane reductions.
// ---------------------------------------------------------------------------

typedef __attribute__((ext_vector_type(16))) __bf16 v16bf;
typedef __attribute__((ext_vector_type(8)))  __bf16 v8bf;
typedef __attribute__((ext_vector_type(8)))  float  v8f;
typedef __attribute__((ext_vector_type(4)))  int    v4i;

#define S_LEN   2048
#define D_MODEL 1024
#define NHEADS  16
#define DKH     64
#define HALF_W  128
#define SCALE_Q 0.125f    // 1/sqrt(64)

static constexpr int WIN = 336;   // per-block key window (64 queries + 2*128 band, padded to 16)
static constexpr int VS  = 344;   // Vt LDS row stride (multiple of 8 -> 16B aligned v8bf)

// ---- async global->LDS (ASYNCcnt) path, guarded: falls back to VGPR copies --
#if defined(__has_builtin)
#  if __has_builtin(__builtin_amdgcn_global_load_async_to_lds_b128) && \
      __has_builtin(__builtin_amdgcn_s_wait_asynccnt)
#    define HAVE_ASYNC_LDS 1
#  endif
#endif
#ifndef HAVE_ASYNC_LDS
#  define HAVE_ASYNC_LDS 0
#endif

__device__ __forceinline__ void async_copy_b128(const void* g, void* l) {
#if HAVE_ASYNC_LDS
  __builtin_amdgcn_global_load_async_to_lds_b128(
      (__attribute__((address_space(1))) v4i*)g,
      (__attribute__((address_space(3))) v4i*)l,
      0 /*imm offset*/, 0 /*cpol*/);
#else
  *(v8bf*)l = *(const v8bf*)g;   // fallback: stage through VGPRs
#endif
}

__device__ __forceinline__ void async_wait0() {
#if HAVE_ASYNC_LDS
  __builtin_amdgcn_s_wait_asynccnt(0);
#endif
}

__device__ __forceinline__ __bf16 f2bf(float f) {
  unsigned int u = __builtin_bit_cast(unsigned int, f);
  unsigned int r = (u + 0x7FFFu + ((u >> 16) & 1u)) >> 16;
  return __builtin_bit_cast(__bf16, (unsigned short)r);
}

__device__ __forceinline__ v8f zero8() {
  v8f z;
#pragma unroll
  for (int i = 0; i < 8; ++i) z[i] = 0.0f;
  return z;
}

// Load a 16x32 bf16 A-fragment from row-major data (row stride in elements).
// Layout used here (contiguous k): lanes 0-15 row=lane, K = 0..15; lanes 16-31 row=lane-16, K = 16..31.
__device__ __forceinline__ v16bf load_a_frag(const __bf16* base, int stride, int lane) {
  const int m  = lane & 15;
  const int kb = (lane >> 4) << 4;     // 0 or 16
  const __bf16* p = base + (size_t)m * stride;
  union { v16bf v; v8bf h[2]; } u;
  u.h[0] = *(const v8bf*)(p + kb);
  u.h[1] = *(const v8bf*)(p + kb + 8);
  return u.v;
}

// Load a 32x16 bf16 B-fragment where B[kk][n] is stored with n as rows of
// `base` (row stride in elements) and kk contiguous along the row.
// Layout used here (contiguous k, as A): lane column n=lane&15; lanes 0-15 hold K=0..15, lanes 16-31 K=16..31.
__device__ __forceinline__ v16bf load_b_frag(const __bf16* base, int stride, int lane) {
  const int n  = lane & 15;
  const int kb = (lane >> 4) << 4;     // 0 or 16
  const __bf16* p = base + (size_t)n * stride + kb;
  union { v16bf v; v8bf h[2]; } u;
  u.h[0] = *(const v8bf*)(p);
  u.h[1] = *(const v8bf*)(p + 8);
  return u.v;
}

// ---------------------------------------------------------------------------
// GEMM: C[m,n] = sum_k A[m,k] * W[n,k] + bias[n]
// MODE 0: A = f32 [4096,1024], C = bf16 in [B,H,S,dk] head layout, scaled.
// MODE 1: A = bf16 [4096,1024] (async-staged), C = f32 [4096,1024].
// Block: 256 threads (8 waves), tile 128x128, k-slab 32.
// ---------------------------------------------------------------------------
template <int MODE>
__global__ __launch_bounds__(256) void gemm_kernel(const void* __restrict__ Ap,
                                                   const float* __restrict__ Wt,
                                                   const float* __restrict__ bias,
                                                   void* __restrict__ Cp,
                                                   float scale) {
  __shared__ __bf16 As[128 * 32];
  __shared__ __bf16 Bs[128 * 32];

  const int tid   = threadIdx.x;
  const int lane  = tid & 31;
  const int wid   = tid >> 5;
  const int wm    = wid >> 1;          // 0..3 (M direction)
  const int wn    = wid & 1;           // 0..1 (N direction)
  const int half  = lane >> 4;
  const int nlane = lane & 15;
  const int mb    = blockIdx.y * 128;
  const int nb    = blockIdx.x * 128;
  const int Kd    = D_MODEL;

  v8f acc[2][4];
#pragma unroll
  for (int mf = 0; mf < 2; ++mf)
#pragma unroll
    for (int nf = 0; nf < 4; ++nf) acc[mf][nf] = zero8();

  for (int kk = 0; kk < Kd; kk += 32) {
    // ---- stage A tile ----
    if (MODE == 0) {
      const float* A = (const float*)Ap;
#pragma unroll
      for (int it = 0; it < 4; ++it) {
        int q   = tid + it * 256;           // quad id, 128*32/4 = 1024
        int row = q >> 3;
        int c4  = (q & 7) * 4;
        float4 f = *(const float4*)(A + (size_t)(mb + row) * Kd + kk + c4);
        __bf16* dst = As + row * 32 + c4;
        dst[0] = f2bf(f.x); dst[1] = f2bf(f.y); dst[2] = f2bf(f.z); dst[3] = f2bf(f.w);
      }
    } else {
      const __bf16* A = (const __bf16*)Ap;
#pragma unroll
      for (int it = 0; it < 2; ++it) {
        int o   = tid + it * 256;           // oct id, 128*32/8 = 512
        int row = o >> 2;
        int c8  = (o & 3) * 8;
        async_copy_b128(A + (size_t)(mb + row) * Kd + kk + c8, As + row * 32 + c8);
      }
    }
    // ---- stage W tile (rows are output columns n), f32 -> bf16 ----
#pragma unroll
    for (int it = 0; it < 4; ++it) {
      int q   = tid + it * 256;
      int row = q >> 3;
      int c4  = (q & 7) * 4;
      float4 f = *(const float4*)(Wt + (size_t)(nb + row) * Kd + kk + c4);
      __bf16* dst = Bs + row * 32 + c4;
      dst[0] = f2bf(f.x); dst[1] = f2bf(f.y); dst[2] = f2bf(f.z); dst[3] = f2bf(f.w);
    }
    if (kk + 32 < Kd)  // hint next W slab into L2 (global_prefetch_b8)
      __builtin_prefetch((const void*)(Wt + (size_t)(nb + (tid >> 1)) * Kd + kk + 32), 0, 1);
    async_wait0();
    __syncthreads();

    // ---- compute: each wave 2x4 fragments ----
    v16bf afr[2];
#pragma unroll
    for (int mf = 0; mf < 2; ++mf)
      afr[mf] = load_a_frag(As + (wm * 32 + mf * 16) * 32, 32, lane);
#pragma unroll
    for (int nf = 0; nf < 4; ++nf) {
      v16bf bfr = load_b_frag(Bs + (wn * 64 + nf * 16) * 32, 32, lane);
#pragma unroll
      for (int mf = 0; mf < 2; ++mf)
        acc[mf][nf] = __builtin_amdgcn_wmma_f32_16x16x32_bf16(
            false, afr[mf], false, bfr, (short)0, acc[mf][nf], false, false);
    }
    __syncthreads();
  }

  // ---- epilogue ----
#pragma unroll
  for (int mf = 0; mf < 2; ++mf)
#pragma unroll
    for (int nf = 0; nf < 4; ++nf)
#pragma unroll
      for (int r = 0; r < 8; ++r) {
        int m = mb + wm * 32 + mf * 16 + r + half * 8;
        int c = nb + wn * 64 + nf * 16 + nlane;
        float v = acc[mf][nf][r] + bias[c];
        if (MODE == 0) {
          v *= scale;
          // [b][h][s][dk]: b = m>>11, s = m&2047, h = c>>6, d = c&63
          size_t idx = ((size_t)((m >> 11) * NHEADS + (c >> 6)) * S_LEN +
                        (size_t)(m & (S_LEN - 1))) * DKH + (c & 63);
          ((__bf16*)Cp)[idx] = f2bf(v);
        } else {
          ((float*)Cp)[(size_t)m * D_MODEL + c] = v;
        }
      }
}

// ---------------------------------------------------------------------------
// Banded flash attention. Grid: (S/64, B*H). Block: 128 threads = 4 waves.
// Wave w owns 16 query rows, streams its 288-key band as 9 pairs of 16-tiles.
// ---------------------------------------------------------------------------
__global__ __launch_bounds__(128) void attn_kernel(const __bf16* __restrict__ Q,
                                                   const __bf16* __restrict__ K,
                                                   const __bf16* __restrict__ V,
                                                   __bf16* __restrict__ AO) {
  __shared__ __bf16 Vts[DKH * VS];        // V window, transposed: [d][key]
  __shared__ __bf16 Vrow[64 * DKH];       // async staging chunk, row-major
  __shared__ __bf16 Ps[4 * 16 * 32];      // per-wave P staging (D-layout -> A-layout)

  const int tid   = threadIdx.x;
  const int lane  = tid & 31;
  const int w     = tid >> 5;
  const int half  = lane >> 4;
  const int nlane = lane & 15;
  const int q0    = blockIdx.x * 64;
  const int bh    = blockIdx.y;           // b*16 + h
  const int kwin0 = q0 - HALF_W;          // global key index of window column 0

  const __bf16* Qbh = Q + (size_t)bh * S_LEN * DKH;
  const __bf16* Kbh = K + (size_t)bh * S_LEN * DKH;
  const __bf16* Vbh = V + (size_t)bh * S_LEN * DKH;

  // ---- stage V window: async copy row-major chunks, then transpose ----
  // OOB keys are address-clamped (not zeroed): their softmax weights are
  // exactly zero (exp of (-1e30 - max) underflows), so values are don't-care.
  for (int chunk = 0; chunk < WIN; chunk += 64) {
    const int nkeys = (WIN - chunk) < 64 ? (WIN - chunk) : 64;   // 64,...,64,16
    for (int idx = tid * 8; idx < nkeys * DKH; idx += 128 * 8) {
      int key = idx >> 6;
      int d0  = idx & 63;
      int gk  = kwin0 + chunk + key;
      gk = gk < 0 ? 0 : (gk > S_LEN - 1 ? S_LEN - 1 : gk);
      async_copy_b128(Vbh + (size_t)gk * DKH + d0, Vrow + key * DKH + d0);
    }
    async_wait0();
    __syncthreads();
    for (int idx = tid * 8; idx < nkeys * DKH; idx += 128 * 8) {
      int key = idx >> 6;
      int d0  = idx & 63;
      v8bf v = *(const v8bf*)(Vrow + key * DKH + d0);
#pragma unroll
      for (int j = 0; j < 8; ++j) Vts[(d0 + j) * VS + (chunk + key)] = v[j];
    }
    __syncthreads();
  }

  const int qrow0 = q0 + 16 * w;

  // Q fragments for this wave's 16 rows (dk=64 -> two k=32 chunks), from global.
  v16bf aq[2];
#pragma unroll
  for (int c = 0; c < 2; ++c)
    aq[c] = load_a_frag(Qbh + (size_t)qrow0 * DKH + c * 32, DKH, lane);

  float mrow[8], lsum[8];
  v8f   oacc[4];
#pragma unroll
  for (int r = 0; r < 8; ++r) { mrow[r] = -3.0e38f; lsum[r] = 0.0f; }
#pragma unroll
  for (int nf = 0; nf < 4; ++nf) oacc[nf] = zero8();

  __bf16* Pw = Ps + w * 16 * 32;

  for (int pair = 0; pair < 9; ++pair) {
    const int krel0 = 16 * w + 32 * pair;          // window-relative first key of pair

    // ---- scores: two 16x16 tiles ----
    v8f sc[2];
#pragma unroll
    for (int t = 0; t < 2; ++t) {
      int kg = kwin0 + krel0 + 16 * t + nlane;     // this lane's key column
      int kc = kg < 0 ? 0 : (kg > S_LEN - 1 ? S_LEN - 1 : kg);
      v8f s = zero8();
#pragma unroll
      for (int c = 0; c < 2; ++c) {
        // B[kk][n] = K[key_n][c*32 + kk] : contiguous 16 bf16 per lane-half
        union { v16bf v; v8bf h[2]; } ub;
        const __bf16* p = Kbh + (size_t)kc * DKH + c * 32 + half * 16;
        ub.h[0] = *(const v8bf*)(p);
        ub.h[1] = *(const v8bf*)(p + 8);
        s = __builtin_amdgcn_wmma_f32_16x16x32_bf16(
            false, aq[c], false, ub.v, (short)0, s, false, false);
      }
      sc[t] = s;
    }

    // ---- mask + online softmax (scale already folded into Q) ----
#pragma unroll
    for (int r = 0; r < 8; ++r) {
      const int m = r + 8 * half;
      const int i = qrow0 + m;
#pragma unroll
      for (int t = 0; t < 2; ++t) {
        int j = kwin0 + krel0 + 16 * t + nlane;
        bool ok = (j >= i - HALF_W) && (j < i + HALF_W) && (j >= 0) && (j < S_LEN);
        sc[t][r] = ok ? sc[t][r] : -1.0e30f;
      }
      float v = fmaxf(sc[0][r], sc[1][r]);
      v = fmaxf(v, __shfl_xor(v, 1, 32));
      v = fmaxf(v, __shfl_xor(v, 2, 32));
      v = fmaxf(v, __shfl_xor(v, 4, 32));
      v = fmaxf(v, __shfl_xor(v, 8, 32));          // row max over 16 lanes of this half
      float newm = fmaxf(mrow[r], v);
      float corr = __expf(mrow[r] - newm);
      mrow[r] = newm;
      float p0 = __expf(sc[0][r] - newm);
      float p1 = __expf(sc[1][r] - newm);
      sc[0][r] = p0; sc[1][r] = p1;
      float rs = p0 + p1;
      rs += __shfl_xor(rs, 1, 32);
      rs += __shfl_xor(rs, 2, 32);
      rs += __shfl_xor(rs, 4, 32);
      rs += __shfl_xor(rs, 8, 32);
      lsum[r] = lsum[r] * corr + rs;
#pragma unroll
      for (int nf = 0; nf < 4; ++nf) oacc[nf][r] *= corr;
    }

    // ---- stage P (D-layout f32 -> A-layout bf16 via wave-private LDS) ----
#pragma unroll
    for (int r = 0; r < 8; ++r) {
      const int m = r + 8 * half;
      Pw[m * 32 + nlane]      = f2bf(sc[0][r]);
      Pw[m * 32 + 16 + nlane] = f2bf(sc[1][r]);
    }
    v16bf ap = load_a_frag(Pw, 32, lane);

    // ---- O += P @ V  (B[kk][n] = V[key][d] = Vts[d][key], contiguous) ----
#pragma unroll
    for (int nf = 0; nf < 4; ++nf) {
      union { v16bf v; v8bf h[2]; } ub;
      const __bf16* p = Vts + (size_t)(nf * 16 + nlane) * VS + krel0 + half * 16;
      ub.h[0] = *(const v8bf*)(p);
      ub.h[1] = *(const v8bf*)(p + 8);
      oacc[nf] = __builtin_amdgcn_wmma_f32_16x16x32_bf16(
          false, ap, false, ub.v, (short)0, oacc[nf], false, false);
    }
  }

  // ---- normalize and write attention output (bf16, concat-head layout) ----
  const int b = bh >> 4, h = bh & 15;
#pragma unroll
  for (int nf = 0; nf < 4; ++nf)
#pragma unroll
    for (int r = 0; r < 8; ++r) {
      const int m   = r + 8 * half;
      const int row = b * S_LEN + qrow0 + m;
      const int col = h * DKH + nf * 16 + nlane;
      AO[(size_t)row * D_MODEL + col] = f2bf(oacc[nf][r] / lsum[r]);
    }
}

// ---------------------------------------------------------------------------
extern "C" void kernel_launch(void* const* d_in, const int* in_sizes, int n_in,
                              void* d_out, int out_size, void* d_ws, size_t ws_size,
                              hipStream_t stream) {
  const float* q_in = (const float*)d_in[0];
  const float* k_in = (const float*)d_in[1];
  const float* v_in = (const float*)d_in[2];
  const float* Wq = (const float*)d_in[3];
  const float* bq = (const float*)d_in[4];
  const float* Wk = (const float*)d_in[5];
  const float* bk = (const float*)d_in[6];
  const float* Wv = (const float*)d_in[7];
  const float* bv = (const float*)d_in[8];
  const float* Wo = (const float*)d_in[9];
  const float* bo = (const float*)d_in[10];

  const size_t qkvElems = (size_t)2 * S_LEN * D_MODEL;  // 4096*1024
  __bf16* Qw = (__bf16*)d_ws;
  __bf16* Kw = Qw + qkvElems;
  __bf16* Vw = Kw + qkvElems;
  __bf16* AO = Vw + qkvElems;   // total 4 * 8 MB = 32 MB of workspace

  dim3 gblk(D_MODEL / 128, (2 * S_LEN) / 128);   // 8 x 32
  dim3 tblk(256);

  gemm_kernel<0><<<gblk, tblk, 0, stream>>>((const void*)q_in, Wq, bq, (void*)Qw, SCALE_Q);
  gemm_kernel<0><<<gblk, tblk, 0, stream>>>((const void*)k_in, Wk, bk, (void*)Kw, 1.0f);
  gemm_kernel<0><<<gblk, tblk, 0, stream>>>((const void*)v_in, Wv, bv, (void*)Vw, 1.0f);

  attn_kernel<<<dim3(S_LEN / 64, 2 * NHEADS), 128, 0, stream>>>(Qw, Kw, Vw, AO);

  gemm_kernel<1><<<gblk, tblk, 0, stream>>>((const void*)AO, Wo, bo, d_out, 1.0f);
}
